// HirAttNeck_66967130079492
// MI455X (gfx1250) — matrix operand order repaired and checked by reading, not compiled, not run on hardware
//
#include <hip/hip_runtime.h>

typedef __attribute__((ext_vector_type(16))) __bf16 v16bf;
typedef __attribute__((ext_vector_type(8)))  float  v8f;

#define QK_SCALE 0.125f  // 64^-0.5

__device__ __forceinline__ __bf16 f2bf(float f) {
    unsigned int u = __builtin_bit_cast(unsigned int, f);
    u += 0x7FFFu + ((u >> 16) & 1u);              // round-to-nearest-even
    unsigned short h = (unsigned short)(u >> 16);
    return __builtin_bit_cast(__bf16, h);
}

// ---------------------------------------------------------------------------
// One-time: WT[m][c][k] = bf16(W[m][k][c]) for the 4 Wp matrices.
// LDS-tiled transpose (64x65 pad), both global sides coalesced.
// Grid: 4 matrices * 64 tiles. Block: 256 threads.
// ---------------------------------------------------------------------------
__global__ void __launch_bounds__(256) transpose_bf16_kernel(const float* __restrict__ W,  // (4,512,512)
                                                             __bf16* __restrict__ WT) {    // (4,512,512)
    __shared__ float ts[64][65];
    const int m    = blockIdx.x >> 6;
    const int tile = blockIdx.x & 63;
    const int rb   = (tile >> 3) * 64;        // k base
    const int cb   = (tile & 7) * 64;         // c base
    const int tx   = threadIdx.x & 63, ty = threadIdx.x >> 6;
    const float* Wm  = W  + (size_t)m * 512 * 512;
    __bf16*      WTm = WT + (size_t)m * 512 * 512;
#pragma unroll 4
    for (int p = 0; p < 16; ++p) {
        const int r = ty + p * 4;
        ts[r][tx] = Wm[(size_t)(rb + r) * 512 + cb + tx];
    }
    __syncthreads();
#pragma unroll 4
    for (int p = 0; p < 16; ++p) {
        const int r = ty + p * 4;
        WTm[(size_t)(cb + r) * 512 + rb + tx] = f2bf(ts[tx][r]);
    }
}

// ---------------------------------------------------------------------------
// out[b,:] = (mean_n x4[b,n,:]) @ Wpq + bpq          (32 x 256 x 512, tiny)
// ---------------------------------------------------------------------------
__global__ void __launch_bounds__(256) pq_kernel(const float* __restrict__ x4,
                                                 const float* __restrict__ Wpq,
                                                 const float* __restrict__ bpq,
                                                 float* __restrict__ out) {
    __shared__ float qm[256];
    const int b = blockIdx.x, t = threadIdx.x;
    const float* xb = x4 + (size_t)b * 49 * 256;
    float s = 0.f;
    for (int n = 0; n < 49; ++n) s += xb[n * 256 + t];
    qm[t] = s * (1.0f / 49.0f);
    __syncthreads();
    for (int col = t; col < 512; col += 256) {
        float acc = bpq[col];
        for (int k = 0; k < 256; ++k) acc += qm[k] * Wpq[(size_t)k * 512 + col];
        out[(size_t)b * 512 + col] = acc;
    }
}

// ---------------------------------------------------------------------------
// WMMA GEMM + residual: dst(32x512) = res + A(32x512) @ W + bias
// W pre-transposed/pre-converted: WT[c][k] bf16 -> B-fragment is one aligned
// 32B vector load. A staged once to LDS as bf16 -> two ds_load_b128 per frag.
// Grid: 8 blocks (64 cols). Block: 8 wave32s; wave -> (row-tile, col-tile).
// ---------------------------------------------------------------------------
__global__ void __launch_bounds__(256) gemm32_kernel(const float* __restrict__ Arow, // (32,512)
                                                     const __bf16* __restrict__ WT,  // (512,512) [c][k]
                                                     const float* __restrict__ bias, // (512,)
                                                     const float* __restrict__ res,  // (32,512)
                                                     float* __restrict__ dst) {      // (32,512)
    __shared__ __bf16 As[32 * 512];
    const int t = threadIdx.x;
    const int w = t >> 5, lane = t & 31, ln16 = lane & 15, hi = lane >> 4;

    for (int e = t; e < 32 * 512; e += 256) As[e] = f2bf(Arow[e]);
    __syncthreads();

    const int mt   = w & 1;
    const int ncol = blockIdx.x * 64 + (w >> 1) * 16 + ln16;
    const __bf16* wcol = WT + (size_t)ncol * 512;

    v8f acc = {};
#pragma unroll
    for (int kk = 0; kk < 16; ++kk) {
        // A 16x32 bf16: lane row = lane%16; elem e -> K = hi*16+e (contiguous).
        v16bf a;
        const __bf16* ar = &As[(mt * 16 + ln16) * 512 + kk * 32 + hi * 16];
#pragma unroll
        for (int e = 0; e < 8; ++e) { a[e] = ar[e]; a[e + 8] = ar[8 + e]; }
        // B 32x16 bf16: lane col = lane%16; elem e -> K = hi*16+e (contiguous).
        const v16bf bfr = *(const v16bf*)(wcol + kk * 32 + hi * 16);
        acc = __builtin_amdgcn_wmma_f32_16x16x32_bf16(
                  false, a, false, bfr, (short)0, acc, false, false);
    }

    // D: VGPR j, lanes 0-15 -> M=j, lanes 16-31 -> M=j+8.
    const float bs = bias[ncol];
#pragma unroll
    for (int j = 0; j < 8; ++j) {
        const int row = mt * 16 + hi * 8 + j;
        dst[(size_t)row * 512 + ncol] = acc[j] + bs + res[(size_t)row * 512 + ncol];
    }
}

// ---------------------------------------------------------------------------
// Fused per-(b,h): q-projection + attention with both kv-projections folded:
//   q_h[d]  = out[b]·Wq[:,h*64+d] + bq        (fused, saves a launch)
//   weff[c] = Wk[c,:]·(q_h*scale)             (k-projection collapsed; k-bias
//                                              shifts all scores -> cancels)
//   s[n]    = x[b,n,:]·weff                   (streaming + prefetch)
//   softmax over n (block-wide)
//   xp[c]   = sum_n attn[n]*x[b,n,c]
//   o[h,d]  = xp·Wv[:,h*64+d] + bv            (v-projection collapsed; sum=1)
// Memory-bound; x streamed twice, second pass from 192MB L2.
// ---------------------------------------------------------------------------
template <int C_IN>
__global__ void __launch_bounds__(256) attn_kernel(const float* __restrict__ x,       // (32,N,C_IN)
                                                   const float* __restrict__ Wkv,     // (C_IN,1024)
                                                   const float* __restrict__ bkv_row, // (1024,)
                                                   const float* __restrict__ outv,    // (32,512)
                                                   const float* __restrict__ Wq,      // (512,512)
                                                   const float* __restrict__ bq_row,  // (512,)
                                                   float* __restrict__ o,             // (32,512)
                                                   int N) {
    __shared__ float out_sh[512];
    __shared__ float q_sh[64];
    __shared__ float weff[C_IN];
    __shared__ float xp[C_IN];
    __shared__ float red[256];
    __shared__ float p_sh[3136];           // max N

    const int b = blockIdx.x >> 3;
    const int h = blockIdx.x & 7;
    const int t = threadIdx.x;

    for (int e = t; e < 512; e += 256) out_sh[e] = outv[(size_t)b * 512 + e];
    __syncthreads();

    // fused q-projection (4 k-stripes x 64 d)
    {
        const int d = t & 63, stripe = t >> 6;
        const float* wq = Wq + h * 64 + d;
        float s = 0.f;
        for (int k = stripe * 128; k < stripe * 128 + 128; ++k)
            s += out_sh[k] * wq[(size_t)k * 512];
        red[t] = s;
    }
    __syncthreads();
    if (t < 64)
        q_sh[t] = (red[t] + red[t + 64] + red[t + 128] + red[t + 192]
                   + bq_row[h * 64 + t]) * QK_SCALE;
    __syncthreads();

    // effective score weights: weff[c] = sum_d Wk[c, h*64+d] * q_sh[d]
    for (int c = t; c < C_IN; c += 256) {
        const float* wr = &Wkv[(size_t)c * 1024 + h * 64];
        float s = 0.f;
#pragma unroll 8
        for (int d = 0; d < 64; ++d) s += wr[d] * q_sh[d];
        weff[c] = s;
    }
    __syncthreads();

    // ---- pass 1: scores + local max ----
    const float* xb = x + (size_t)b * N * C_IN;
    float lmax = -3.0e38f;
    for (int n = t; n < N; n += 256) {
        const float* xr = xb + (size_t)n * C_IN;
        __builtin_prefetch(xr + 256 * C_IN, 0, 1);     // global_prefetch_b8
        float s = 0.f;
#pragma unroll
        for (int c = 0; c < C_IN; c += 4) {
            const float4 xv = *(const float4*)(xr + c);
            s += xv.x * weff[c] + xv.y * weff[c + 1] +
                 xv.z * weff[c + 2] + xv.w * weff[c + 3];
        }
        p_sh[n] = s;
        lmax = fmaxf(lmax, s);
    }
    red[t] = lmax;
    __syncthreads();
    for (int s = 128; s > 0; s >>= 1) {
        if (t < s) red[t] = fmaxf(red[t], red[t + s]);
        __syncthreads();
    }
    const float M = red[0];
    __syncthreads();

    // ---- exp + local sum ----
    float lsum = 0.f;
    for (int n = t; n < N; n += 256) {
        const float p = __expf(p_sh[n] - M);
        p_sh[n] = p;
        lsum += p;
    }
    red[t] = lsum;
    __syncthreads();
    for (int s = 128; s > 0; s >>= 1) {
        if (t < s) red[t] += red[t + s];
        __syncthreads();
    }
    const float Sinv = 1.0f / red[0];
    __syncthreads();

    // ---- pass 2: xp[c] = sum_n p[n]*x[n,c]  (c fast across lanes: coalesced) ----
    constexpr int TPC = 256 / C_IN;
    const int col    = t % C_IN;
    const int stripe = t / C_IN;
    float accx = 0.f;
    for (int n = stripe; n < N; n += TPC)
        accx += p_sh[n] * xb[(size_t)n * C_IN + col];
    red[t] = accx;
    __syncthreads();
    if (t < C_IN) {
        float s = 0.f;
#pragma unroll
        for (int st = 0; st < TPC; ++st) s += red[st * C_IN + t];
        xp[t] = s * Sinv;
    }
    __syncthreads();

    // ---- folded v-projection ----
    if (t < 64) {
        const int colg = 512 + h * 64 + t;
        float s = bkv_row[colg];
        for (int c = 0; c < C_IN; ++c) s += xp[c] * Wkv[(size_t)c * 1024 + colg];
        o[(size_t)b * 512 + h * 64 + t] = s;
    }
}

// ---------------------------------------------------------------------------
extern "C" void kernel_launch(void* const* d_in, const int* in_sizes, int n_in,
                              void* d_out, int out_size, void* d_ws, size_t ws_size,
                              hipStream_t stream) {
    (void)in_sizes; (void)n_in; (void)out_size; (void)ws_size;
    const float* x[4]   = {(const float*)d_in[0], (const float*)d_in[1],
                           (const float*)d_in[2], (const float*)d_in[3]};
    const float* Wpq    = (const float*)d_in[4];
    const float* bpq    = (const float*)d_in[5];
    const float* Wq     = (const float*)d_in[6];
    const float* bq     = (const float*)d_in[7];
    const float* Wp     = (const float*)d_in[8];
    const float* bp     = (const float*)d_in[9];
    const float* Wkv[4] = {(const float*)d_in[10], (const float*)d_in[11],
                           (const float*)d_in[12], (const float*)d_in[13]};
    const float* bkv    = (const float*)d_in[14];

    float*  out_buf = (float*)d_ws;              // 32*512 f32
    float*  o_buf   = out_buf + 32 * 512;        // 32*512 f32
    __bf16* WpT     = (__bf16*)(o_buf + 32 * 512);  // 4*512*512 bf16 (2 MB)

    static const int Ns[4] = {3136, 784, 196, 49};

    transpose_bf16_kernel<<<4 * 64, 256, 0, stream>>>(Wp, WpT);
    pq_kernel<<<32, 256, 0, stream>>>(x[3], Wpq, bpq, out_buf);

    for (int i = 3; i >= 0; --i) {
        switch (i) {
            case 0: attn_kernel<32 ><<<256, 256, 0, stream>>>(x[0], Wkv[0], bkv + 0 * 1024, out_buf, Wq + (size_t)0 * 262144, bq + 0 * 512, o_buf, Ns[0]); break;
            case 1: attn_kernel<64 ><<<256, 256, 0, stream>>>(x[1], Wkv[1], bkv + 1 * 1024, out_buf, Wq + (size_t)1 * 262144, bq + 1 * 512, o_buf, Ns[1]); break;
            case 2: attn_kernel<128><<<256, 256, 0, stream>>>(x[2], Wkv[2], bkv + 2 * 1024, out_buf, Wq + (size_t)2 * 262144, bq + 2 * 512, o_buf, Ns[2]); break;
            case 3: attn_kernel<256><<<256, 256, 0, stream>>>(x[3], Wkv[3], bkv + 3 * 1024, out_buf, Wq + (size_t)3 * 262144, bq + 3 * 512, o_buf, Ns[3]); break;
        }
        // out = out + o @ Wp_i + bp_i   (WMMA, pre-transposed bf16 weights)
        float* dst = (i == 0) ? (float*)d_out : out_buf;
        gemm32_kernel<<<8, 256, 0, stream>>>(o_buf, WpT + (size_t)i * 262144,
                                             bp + i * 512, out_buf, dst);
    }
}
